// Decoder_44032004718787
// MI455X (gfx1250) — compile-verified, matrix operand order checked by reading, not run on hardware
//
#include <hip/hip_runtime.h>
#include <hip/hip_bf16.h>
#include <math.h>

// ---------------------------------------------------------------------------
// Dimensions (fixed by the reference)
// ---------------------------------------------------------------------------
#define BB   32      // batch
#define LXX  64      // source length
#define TT   64      // target length
#define DD   1024    // hidden
#define DWW  512     // embedding width
#define VV   32000   // vocab
#define KCAT 3584    // DW + 2D + D  (xcat = [emb | feed | h])
#define KCR  3072    // 3D           (zcat = [h | ctx])
#define K2D  2048    // 2D

typedef __attribute__((ext_vector_type(16))) __bf16 v16bf;
typedef __attribute__((ext_vector_type(8)))  __bf16 v8bf;
typedef __attribute__((ext_vector_type(8)))  float  v8f;

__device__ __forceinline__ float sigm(float x) { return 1.0f / (1.0f + __expf(-x)); }

// ---------------------------------------------------------------------------
// Generic skinny GEMM:  C[M,N] = A[M,K] * W[N,K]^T   (A,W bf16; acc f32)
// One wave computes a 32-row x 16-col tile via two V_WMMA_F32_16X16X32_BF16.
// blockDim = 256 (8 waves), grid.x = N/128, grid.y = M/32.
//
// Fragment layout per CDNA5 ISA 7.12.2 (16-bit A, 16x32):
//   lane L (L<16):  row = L,    elems 0..7 = K[k..k+7],    8..15 = K[k+16..k+23]
//   lane L (L>=16): row = L-16, elems 0..7 = K[k+8..k+15], 8..15 = K[k+24..k+31]
// B (32x16, K-major) mirrors this with lane -> output column n, and
// B[k][n] = W[n][k], so each lane streams weight row n contiguously.
// C/D: VGPR j, lane L  ->  row = (L>>4)*8 + j, col = L&15.
//
// EPI: 0 = store f32 to Of (ldo)            (gates, k_enc, logits)
//      1 = store bf16(tanh(x)) to Obf       (readout 'pre')
//      2 = Of = tanh(x), Of2 = x            (bridge: h0 and c0)
// ---------------------------------------------------------------------------
template <int EPI>
__global__ __launch_bounds__(256) void wmma_gemm_k(
    const __bf16* __restrict__ A, int lda,
    const __bf16* __restrict__ W, int ldw,
    float* __restrict__ Of, long ldo,
    float* __restrict__ Of2,
    __bf16* __restrict__ Obf, int ldob,
    int K) {
  const int lane = threadIdx.x & 31;
  const int wave = threadIdx.x >> 5;
  const int col0 = (blockIdx.x * 8 + wave) * 16;
  const int row0 = blockIdx.y * 32;
  const int r    = lane & 15;
  const int hi   = lane >> 4;

  const __bf16* a0 = A + (long)(row0 + r) * lda + hi * 8;
  const __bf16* a1 = A + (long)(row0 + 16 + r) * lda + hi * 8;
  const __bf16* bp = W + (long)(col0 + r) * ldw + hi * 8;

  v8f acc0 = {};
  v8f acc1 = {};
  for (int k = 0; k < K; k += 32) {
    __builtin_prefetch(bp + k + 128, 0, 1);  // stream next weight chunk (global_prefetch)
    v8bf al0 = *(const v8bf*)(a0 + k);
    v8bf ah0 = *(const v8bf*)(a0 + k + 16);
    v8bf al1 = *(const v8bf*)(a1 + k);
    v8bf ah1 = *(const v8bf*)(a1 + k + 16);
    v8bf bl  = *(const v8bf*)(bp + k);
    v8bf bh  = *(const v8bf*)(bp + k + 16);
    v16bf fa0 = __builtin_shufflevector(al0, ah0, 0, 1, 2, 3, 4, 5, 6, 7, 8, 9, 10, 11, 12, 13, 14, 15);
    v16bf fa1 = __builtin_shufflevector(al1, ah1, 0, 1, 2, 3, 4, 5, 6, 7, 8, 9, 10, 11, 12, 13, 14, 15);
    v16bf fb  = __builtin_shufflevector(bl, bh, 0, 1, 2, 3, 4, 5, 6, 7, 8, 9, 10, 11, 12, 13, 14, 15);
    acc0 = __builtin_amdgcn_wmma_f32_16x16x32_bf16(false, fa0, false, fb, (short)0, acc0, false, false);
    acc1 = __builtin_amdgcn_wmma_f32_16x16x32_bf16(false, fa1, false, fb, (short)0, acc1, false, false);
  }

  const int colg = col0 + r;
#pragma unroll
  for (int j = 0; j < 8; ++j) {
    const int r0 = row0 + hi * 8 + j;
    const int r1 = r0 + 16;
    if (EPI == 0) {
      Of[(long)r0 * ldo + colg] = acc0[j];
      Of[(long)r1 * ldo + colg] = acc1[j];
    } else if (EPI == 1) {
      Obf[r0 * ldob + colg] = (__bf16)tanhf(acc0[j]);
      Obf[r1 * ldob + colg] = (__bf16)tanhf(acc1[j]);
    } else {
      float v0 = acc0[j], v1 = acc1[j];
      Of2[(long)r0 * ldo + colg] = v0;          // c0
      Of2[(long)r1 * ldo + colg] = v1;
      Of[(long)r0 * ldo + colg] = tanhf(v0);    // h0
      Of[(long)r1 * ldo + colg] = tanhf(v1);
    }
  }
}

// ---------------------------------------------------------------------------
// Setup / elementwise kernels
// ---------------------------------------------------------------------------
__global__ void f32_to_bf16_k(const float* __restrict__ s, __bf16* __restrict__ d, long n) {
  long i = (long)blockIdx.x * blockDim.x + threadIdx.x;
  long st = (long)gridDim.x * blockDim.x;
  for (; i < n; i += st) d[i] = (__bf16)s[i];
}

// Wcat[n, 0:2560] = W_ih[n, :], Wcat[n, 2560:3584] = W_hh[n, :]
__global__ void build_wcat_k(const float* __restrict__ wih, const float* __restrict__ whh,
                             __bf16* __restrict__ out) {
  long i = (long)blockIdx.x * blockDim.x + threadIdx.x;
  long st = (long)gridDim.x * blockDim.x;
  const long n = (long)(4 * DD) * KCAT;
  for (; i < n; i += st) {
    long row = i / KCAT;
    long k = i % KCAT;
    float v = (k < (DWW + K2D)) ? wih[row * (DWW + K2D) + k] : whh[row * DD + (k - (DWW + K2D))];
    out[i] = (__bf16)v;
  }
}

// xct_cat[b, :] = [x_ct[0,b,:], x_ct[1,b,:]]  (bf16)
__global__ void build_xct_k(const float* __restrict__ xct, __bf16* __restrict__ out) {
  int i = blockIdx.x * blockDim.x + threadIdx.x;  // 32*2048
  int b = i / K2D, k = i % K2D;
  int s = (k < DD) ? 0 : 1;
  out[i] = (__bf16)xct[(long)s * BB * DD + (long)b * DD + (k & (DD - 1))];
}

__global__ void zero_feed_k(float* __restrict__ feed) {
  int i = blockIdx.x * blockDim.x + threadIdx.x;  // 32*2048
  feed[i] = 0.0f;
}

// xcat[b, :] = [ emb[y[b,t]], feed[b], h[b] ]   (bf16)
__global__ void build_xcat_k(const float* __restrict__ emb, const int* __restrict__ y,
                             const float* __restrict__ feed, const float* __restrict__ h,
                             __bf16* __restrict__ xcat, int t) {
  int i = blockIdx.x * blockDim.x + threadIdx.x;  // 32*3584
  int b = i / KCAT, k = i % KCAT;
  float v;
  if (k < DWW) {
    int tok = y[b * TT + t];
    v = emb[(long)tok * DWW + k];
  } else if (k < DWW + K2D) {
    v = feed[b * K2D + (k - DWW)];
  } else {
    v = h[b * DD + (k - DWW - K2D)];
  }
  xcat[i] = (__bf16)v;
}

// LSTM pointwise: gates[b, {i,f,g,o}] -> h,c  (torch gate order i,f,g,o)
__global__ void lstm_update_k(const float* __restrict__ gates, const float* __restrict__ bih,
                              const float* __restrict__ bhh, float* __restrict__ h,
                              float* __restrict__ c) {
  int i = blockIdx.x * blockDim.x + threadIdx.x;  // 32*1024
  int b = i >> 10, u = i & (DD - 1);
  const float* g = gates + (long)b * 4 * DD;
  float gi = g[u] + bih[u] + bhh[u];
  float gf = g[DD + u] + bih[DD + u] + bhh[DD + u];
  float gg = g[2 * DD + u] + bih[2 * DD + u] + bhh[2 * DD + u];
  float go = g[3 * DD + u] + bih[3 * DD + u] + bhh[3 * DD + u];
  float ct = sigm(gf) * c[i] + sigm(gi) * tanhf(gg);
  c[i] = ct;
  h[i] = sigm(go) * tanhf(ct);
}

// Attention for one batch row per block: scores -> softmax -> ctx.
// Writes feed (f32 ctx) and zcat = [bf16(h) | bf16(ctx)].
__global__ __launch_bounds__(256) void attention_k(
    const float* __restrict__ kenc, const float* __restrict__ xenc,
    const unsigned char* __restrict__ xmask, const float* __restrict__ h,
    float* __restrict__ feed, __bf16* __restrict__ zcat) {
  const int b = blockIdx.x;
  const int tid = threadIdx.x;
  __shared__ float sh_h[DD];
  __shared__ float sh_w[LXX];
  __shared__ float sh_inv;

  for (int i = tid; i < DD; i += 256) sh_h[i] = h[(long)b * DD + i];
  __syncthreads();

  const int wave = tid >> 5, lane = tid & 31;
  for (int l = wave; l < LXX; l += 8) {
    const float* kr = kenc + ((long)b * LXX + l) * DD;
    float s = 0.0f;
    for (int i = lane; i < DD; i += 32) s += kr[i] * sh_h[i];
#pragma unroll
    for (int off = 16; off > 0; off >>= 1) s += __shfl_down(s, off, 32);
    if (lane == 0) sh_w[l] = xmask[b * LXX + l] ? -1.0e9f : s;
  }
  __syncthreads();

  if (tid == 0) {
    float m = -3.0e38f;
    for (int l = 0; l < LXX; ++l) m = fmaxf(m, sh_w[l]);
    float ssum = 0.0f;
    for (int l = 0; l < LXX; ++l) {
      float e = __expf(sh_w[l] - m);
      sh_w[l] = e;
      ssum += e;
    }
    sh_inv = 1.0f / ssum;
  }
  __syncthreads();
  const float inv = sh_inv;

  for (int v = tid; v < K2D; v += 256) {
    const float* xr = xenc + (long)b * LXX * K2D + v;
    float acc = 0.0f;
#pragma unroll 4
    for (int l = 0; l < LXX; ++l) acc += sh_w[l] * xr[(long)l * K2D];
    acc *= inv;
    feed[b * K2D + v] = acc;
    zcat[b * KCR + DD + v] = (__bf16)acc;
  }
  for (int i = tid; i < DD; i += 256) zcat[b * KCR + i] = (__bf16)sh_h[i];
}

// ---------------------------------------------------------------------------
// Host driver
// ---------------------------------------------------------------------------
static inline size_t alignup(size_t x) { return (x + 255) & ~(size_t)255; }

extern "C" void kernel_launch(void* const* d_in, const int* in_sizes, int n_in,
                              void* d_out, int out_size, void* d_ws, size_t ws_size,
                              hipStream_t stream) {
  (void)in_sizes; (void)n_in; (void)out_size; (void)ws_size;

  const float* x_enc   = (const float*)d_in[0];           // [B,LX,2D]
  const float* x_ct    = (const float*)d_in[2];           // [2,B,D]
  const unsigned char* x_mask = (const unsigned char*)d_in[3];  // [B,LX]
  const int*   y_train = (const int*)d_in[4];             // [B,T]
  const float* W_bridge = (const float*)d_in[6];          // [D,2D]
  const float* W_enc_k  = (const float*)d_in[7];          // [D,2D]
  const float* W_cr     = (const float*)d_in[8];          // [D,3D]
  const float* W_ro     = (const float*)d_in[9];          // [V,D]
  const float* emb      = (const float*)d_in[10];         // [V,DW]
  const float* W_ih     = (const float*)d_in[11];         // [4D,DW+2D]
  const float* W_hh     = (const float*)d_in[12];         // [4D,D]
  const float* b_ih     = (const float*)d_in[13];         // [4D]
  const float* b_hh     = (const float*)d_in[14];         // [4D]
  float* out = (float*)d_out;                             // [B,T,V]

  // ---- workspace carve-up (bf16 weight copies + activations) ----
  char* ws = (char*)d_ws;
  size_t off = 0;
  auto take = [&](size_t bytes) { char* p = ws + off; off = alignup(off + bytes); return p; };

  __bf16* Wcat_bf  = (__bf16*)take((size_t)4 * DD * KCAT * 2);   // 28 MB
  __bf16* Wcr_bf   = (__bf16*)take((size_t)DD * KCR * 2);        // 6 MB
  __bf16* Wro_bf   = (__bf16*)take((size_t)VV * DD * 2);         // 62.5 MB
  __bf16* Wenck_bf = (__bf16*)take((size_t)DD * K2D * 2);        // 4 MB
  __bf16* Wbr_bf   = (__bf16*)take((size_t)DD * K2D * 2);        // 4 MB
  __bf16* xenc_bf  = (__bf16*)take((size_t)BB * LXX * K2D * 2);  // 8 MB
  __bf16* xct_bf   = (__bf16*)take((size_t)BB * K2D * 2);
  float*  kenc_f   = (float*)take((size_t)BB * LXX * DD * 4);    // 8 MB
  float*  h_f      = (float*)take((size_t)BB * DD * 4);
  float*  c_f      = (float*)take((size_t)BB * DD * 4);
  float*  feed_f   = (float*)take((size_t)BB * K2D * 4);
  float*  gates_f  = (float*)take((size_t)BB * 4 * DD * 4);
  __bf16* xcat_bf  = (__bf16*)take((size_t)BB * KCAT * 2);
  __bf16* zcat_bf  = (__bf16*)take((size_t)BB * KCR * 2);
  __bf16* pre_bf   = (__bf16*)take((size_t)BB * DD * 2);

  // ---- one-time setup: bf16 weight copies, bridged state, attention keys ----
  build_wcat_k<<<8192, 256, 0, stream>>>(W_ih, W_hh, Wcat_bf);
  f32_to_bf16_k<<<4096, 256, 0, stream>>>(W_cr, Wcr_bf, (long)DD * KCR);
  f32_to_bf16_k<<<8192, 256, 0, stream>>>(W_ro, Wro_bf, (long)VV * DD);
  f32_to_bf16_k<<<4096, 256, 0, stream>>>(W_enc_k, Wenck_bf, (long)DD * K2D);
  f32_to_bf16_k<<<4096, 256, 0, stream>>>(W_bridge, Wbr_bf, (long)DD * K2D);
  f32_to_bf16_k<<<4096, 256, 0, stream>>>(x_enc, xenc_bf, (long)BB * LXX * K2D);
  build_xct_k<<<(BB * K2D) / 256, 256, 0, stream>>>(x_ct, xct_bf);
  zero_feed_k<<<(BB * K2D) / 256, 256, 0, stream>>>(feed_f);

  // bridge: c0 = xct_cat @ W_bridge^T ; h0 = tanh(c0)    (M=32,N=1024,K=2048)
  wmma_gemm_k<2><<<dim3(DD / 128, 1), 256, 0, stream>>>(
      xct_bf, K2D, Wbr_bf, K2D, h_f, (long)DD, c_f, nullptr, 0, K2D);

  // k_enc = x_enc @ W_enc_k^T                            (M=2048,N=1024,K=2048)
  wmma_gemm_k<0><<<dim3(DD / 128, (BB * LXX) / 32), 256, 0, stream>>>(
      xenc_bf, K2D, Wenck_bf, K2D, kenc_f, (long)DD, nullptr, nullptr, 0, K2D);

  // ---- recurrence over T steps ----
  for (int t = 0; t < TT; ++t) {
    // xcat = [emb[y_t] | feed | h]  (bf16)
    build_xcat_k<<<(BB * KCAT) / 256, 256, 0, stream>>>(emb, y_train, feed_f, h_f, xcat_bf, t);

    // gates = xcat @ [W_ih|W_hh]^T                       (M=32,N=4096,K=3584)
    wmma_gemm_k<0><<<dim3((4 * DD) / 128, 1), 256, 0, stream>>>(
        xcat_bf, KCAT, Wcat_bf, KCAT, gates_f, (long)(4 * DD), nullptr, nullptr, 0, KCAT);

    // LSTM pointwise update -> h, c
    lstm_update_k<<<(BB * DD) / 256, 256, 0, stream>>>(gates_f, b_ih, b_hh, h_f, c_f);

    // attention: scores/softmax/ctx; writes feed and zcat=[h|ctx] bf16
    attention_k<<<BB, 256, 0, stream>>>(kenc_f, x_enc, x_mask, h_f, feed_f, zcat_bf);

    // pre = tanh(zcat @ W_cr^T)  -> bf16                 (M=32,N=1024,K=3072)
    wmma_gemm_k<1><<<dim3(DD / 128, 1), 256, 0, stream>>>(
        zcat_bf, KCR, Wcr_bf, KCR, nullptr, 0, nullptr, pre_bf, DD, KCR);

    // logits_t = pre @ W_ro^T  -> out[:, t, :]           (M=32,N=32000,K=1024)
    wmma_gemm_k<0><<<dim3(VV / 128, 1), 256, 0, stream>>>(
        pre_bf, DD, Wro_bf, DD, out + (long)t * VV, (long)TT * VV, nullptr, nullptr, 0, DD);
  }
}
